// Aligner_55705725829266
// MI455X (gfx1250) — matrix operand order repaired and checked by reading, not compiled, not run on hardware
//
#include <hip/hip_runtime.h>
#include <math.h>

typedef __attribute__((ext_vector_type(16))) _Float16 v16h;
typedef __attribute__((ext_vector_type(8)))  _Float16 v8h;
typedef __attribute__((ext_vector_type(8)))  float    v8f;

#define NB    32
#define TDLEN 1024
#define TELEN 384
#define CPAD  96
#define LRELU 0.3f
#define TEMPC 0.0005f

// ---------------------------------------------------------------------------
// Fragment helpers (16-bit WMMA operands in contiguous k order, same for A and B)
// A 16x32: lane row = lane&15; elem e -> K = k0 + 16*half + e (16 contiguous f16)
__device__ __forceinline__ v16h ldA(const _Float16* p, int half) {
  v8h lo = *(const v8h*)(p + (half << 4));
  v8h hi = *(const v8h*)(p + 8 + (half << 4));
  return __builtin_shufflevector(lo, hi, 0, 1, 2, 3, 4, 5, 6, 7,
                                 8, 9, 10, 11, 12, 13, 14, 15);
}
// B 32x16: lane col = lane&15; elem e -> K = k0 + 16*half + e (16 contiguous f16)
__device__ __forceinline__ v16h ldBrow(const _Float16* p) {
  v8h lo = *(const v8h*)p;
  v8h hi = *(const v8h*)(p + 8);
  return __builtin_shufflevector(lo, hi, 0, 1, 2, 3, 4, 5, 6, 7,
                                 8, 9, 10, 11, 12, 13, 14, 15);
}
#define WMMA(A, B, C) __builtin_amdgcn_wmma_f32_16x16x32_f16(false, A, false, B, \
                                                             (short)0, C, false, false)

// ---------------------------------------------------------------------------
// (B,C,T) f32 -> interior of [b][Tp][CS] f16 (halo rows pre-zeroed by memset)
__global__ void k_transpose16(const float* __restrict__ in, _Float16* __restrict__ out,
                              int C, int T, int CS, int outBatch) {
  int b = blockIdx.y;
  int idx = blockIdx.x * blockDim.x + threadIdx.x;
  if (idx >= C * T) return;
  int c = idx / T, t = idx - c * T;
  out[(size_t)b * outBatch + (size_t)t * CS + c] = (_Float16)in[(size_t)b * C * T + idx];
}

// (Cout,Cin,Kw) f32 -> [Cout][Kw*CinP] f16, K-order k = dk*CinP + ci, zero pad
__global__ void k_prep_weights(const float* __restrict__ w, _Float16* __restrict__ wt,
                               int Cout, int Cin, int CinP, int Kw) {
  int Kpad = Kw * CinP;
  int idx = blockIdx.x * blockDim.x + threadIdx.x;
  if (idx >= Cout * Kpad) return;
  int co = idx / Kpad, k = idx - co * Kpad;
  int dk = k / CinP, ci = k - dk * CinP;
  float v = 0.f;
  if (ci < Cin) v = w[(size_t)co * Cin * Kw + (size_t)ci * Kw + dk];
  wt[idx] = (_Float16)v;
}

// ---------------------------------------------------------------------------
// Big leaky conv1d ('same') with zero-halo rows: one wave = 32(co) x 32(t) tile.
// The (dk,ci) K-order makes the im2col window a CONTIGUOUS slice of [T][CinP],
// so B loads are unguarded b128 with linear advance. 4 WMMAs / 8 loads.
// grid (T/32, B, Cout/32). in/out point at interior row 0.
__global__ __launch_bounds__(32)
void k_conv32(const _Float16* __restrict__ in, const _Float16* __restrict__ wt,
              const float* __restrict__ bias, _Float16* __restrict__ out,
              int CinP, int Kw, int coStride, int inBatch, int outBatch) {
  int Kpad = CinP * Kw;
  int t0  = blockIdx.x << 5;
  int b   = blockIdx.y;
  int co0 = blockIdx.z << 5;
  int lane = threadIdx.x;
  int half = lane >> 4;
  int mn = lane & 15;
  int pad = Kw >> 1;
  const _Float16* pB0 = in + (size_t)b * inBatch +
                        (size_t)(t0 + mn - pad) * CinP + (half << 4);
  const _Float16* pB1 = pB0 + (size_t)16 * CinP;
  const _Float16* w0 = wt + (size_t)(co0 + mn) * Kpad;
  const _Float16* w1 = w0 + (size_t)16 * Kpad;
  v8f acc00 = {}, acc01 = {}, acc10 = {}, acc11 = {};
  for (int k0 = 0; k0 < Kpad; k0 += 32) {
    v16h a0 = ldA(w0 + k0, half);
    v16h a1 = ldA(w1 + k0, half);
    v16h b0 = ldBrow(pB0 + k0);
    v16h b1 = ldBrow(pB1 + k0);
    acc00 = WMMA(a0, b0, acc00);
    acc01 = WMMA(a0, b1, acc01);
    acc10 = WMMA(a1, b0, acc10);
    acc11 = WMMA(a1, b1, acc11);
  }
  _Float16* outB = out + (size_t)b * outBatch;
#pragma unroll
  for (int i = 0; i < 2; ++i) {
#pragma unroll
    for (int j = 0; j < 2; ++j) {
      v8f acc = (i == 0) ? (j == 0 ? acc00 : acc01) : (j == 0 ? acc10 : acc11);
      int coBase = co0 + (i << 4);
      int t = t0 + (j << 4) + mn;
      v8h ov;
#pragma unroll
      for (int r = 0; r < 8; ++r) {
        float v = acc[r] + bias[coBase + r + (half << 3)];
        v = (v >= 0.f) ? v : LRELU * v;
        ov[r] = (_Float16)v;
      }
      *(v8h*)(outB + (size_t)t * coStride + coBase + (half << 3)) = ov;
    }
  }
}

// ---------------------------------------------------------------------------
// k=1 projection conv (Cout=80): one wave = 16x16 tile, no leaky, accumulates
// per-t sum of squares into sq. grid (T/16, B, Cout/16). No halo needed.
__global__ __launch_bounds__(32)
void k_conv16_proj(const _Float16* __restrict__ in, const _Float16* __restrict__ wt,
                   const float* __restrict__ bias, _Float16* __restrict__ out,
                   float* __restrict__ sq, int T, int CinP, int coStride,
                   int inBatch, int outBatch) {
  int t0  = blockIdx.x << 4;
  int b   = blockIdx.y;
  int co0 = blockIdx.z << 4;
  int lane = threadIdx.x;
  int half = lane >> 4;
  int mn = lane & 15;
  const _Float16* wrow = wt + (size_t)(co0 + mn) * CinP;
  const _Float16* brow = in + (size_t)b * inBatch + (size_t)(t0 + mn) * CinP +
                         (half << 4);
  v8f acc = {};
  for (int k0 = 0; k0 < CinP; k0 += 32) {
    v16h a = ldA(wrow + k0, half);
    v16h bm = ldBrow(brow + k0);
    acc = WMMA(a, bm, acc);
  }
  _Float16* outB = out + (size_t)b * outBatch;
  int t = t0 + mn;
  float ss = 0.f;
  v8h ov;
#pragma unroll
  for (int r = 0; r < 8; ++r) {
    float v = acc[r] + bias[co0 + r + (half << 3)];
    _Float16 h = (_Float16)v;
    ov[r] = h;
    float vf = (float)h;
    ss += vf * vf;
  }
  *(v8h*)(outB + (size_t)t * coStride + co0 + (half << 3)) = ov;
  atomicAdd(&sq[(size_t)b * T + t], ss);
}

// ---------------------------------------------------------------------------
// Scores + double log-softmax. One block = (b, 16 TD rows) over full TE.
// q16: [B][TD][96] f16, k16: [B][TE][96] f16 (pad channels zero).
__global__ __launch_bounds__(256)
void k_scores(const _Float16* __restrict__ q16, const _Float16* __restrict__ k16,
              const float* __restrict__ q2, const float* __restrict__ k2,
              const float* __restrict__ prior,     // [B][TE][TD]
              float* __restrict__ logprob, float* __restrict__ soft) {
  __shared__ float xt[16][TELEN];
  __shared__ float red[16][16];
  __shared__ float rowv[16];
  int b = blockIdx.y;
  int td0 = blockIdx.x << 4;
  int tid = threadIdx.x;
  int lane = tid & 31;
  int wv = tid >> 5;
  int half = lane >> 4;
  int mn = lane & 15;
  const _Float16* Q = q16 + (size_t)b * TDLEN * CPAD;
  const _Float16* K = k16 + (size_t)b * TELEN * CPAD;

  v16h aq[3];
#pragma unroll
  for (int ks = 0; ks < 3; ++ks)
    aq[ks] = ldA(Q + (size_t)(td0 + mn) * CPAD + ks * 32, half);
  float q2v[8];
#pragma unroll
  for (int r = 0; r < 8; ++r) q2v[r] = q2[(size_t)b * TDLEN + td0 + r + (half << 3)];

  for (int tt = wv; tt < 24; tt += 8) {
    int te0 = tt << 4;
    v8f acc = {};
#pragma unroll
    for (int ks = 0; ks < 3; ++ks) {
      v16h bm = ldBrow(K + (size_t)(te0 + mn) * CPAD + ks * 32 + (half << 4));
      acc = WMMA(aq[ks], bm, acc);
    }
    float k2v = k2[(size_t)b * TELEN + te0 + mn];
#pragma unroll
    for (int r = 0; r < 8; ++r) {
      int m = r + (half << 3);
      xt[m][te0 + mn] = -TEMPC * (q2v[r] + k2v - 2.f * acc[r]);
    }
  }
  __syncthreads();

  int row = tid & 15, ch = tid >> 4;
  int c0 = ch * 24;
  // pass 1: lse over raw scores
  float m1 = -__builtin_inff();
  for (int c = 0; c < 24; ++c) m1 = fmaxf(m1, xt[row][c0 + c]);
  red[row][ch] = m1; __syncthreads();
  if (tid < 16) { float mm = -__builtin_inff();
    for (int j = 0; j < 16; ++j) mm = fmaxf(mm, red[tid][j]); rowv[tid] = mm; }
  __syncthreads();
  float rm = rowv[row];
  float s = 0.f;
  for (int c = 0; c < 24; ++c) s += expf(xt[row][c0 + c] - rm);
  red[row][ch] = s; __syncthreads();
  if (tid < 16) { float ss = 0.f;
    for (int j = 0; j < 16; ++j) ss += red[tid][j]; rowv[tid] = rowv[tid] + logf(ss); }
  __syncthreads();
  float lse1 = rowv[row];
  __syncthreads();

  // y = x - lse1 + log(prior^T + 1e-8); write logprob
  size_t obase = (size_t)b * TDLEN * TELEN + (size_t)(td0 + row) * TELEN;
  const float* P = prior + (size_t)b * TELEN * TDLEN;
  for (int c = 0; c < 24; ++c) {
    int te = c0 + c;
    float y = xt[row][te] - lse1 + logf(P[(size_t)te * TDLEN + td0 + row] + 1e-8f);
    xt[row][te] = y;
    logprob[obase + te] = y;
  }
  __syncthreads();

  // pass 2: lse over y; soft = exp(y - lse2)
  m1 = -__builtin_inff();
  for (int c = 0; c < 24; ++c) m1 = fmaxf(m1, xt[row][c0 + c]);
  red[row][ch] = m1; __syncthreads();
  if (tid < 16) { float mm = -__builtin_inff();
    for (int j = 0; j < 16; ++j) mm = fmaxf(mm, red[tid][j]); rowv[tid] = mm; }
  __syncthreads();
  rm = rowv[row];
  s = 0.f;
  for (int c = 0; c < 24; ++c) s += expf(xt[row][c0 + c] - rm);
  red[row][ch] = s; __syncthreads();
  if (tid < 16) { float ss = 0.f;
    for (int j = 0; j < 16; ++j) ss += red[tid][j]; rowv[tid] = rowv[tid] + logf(ss); }
  __syncthreads();
  float lse2 = rowv[row];
  for (int c = 0; c < 24; ++c) {
    int te = c0 + c;
    soft[obase + te] = expf(xt[row][te] - lse2);
  }
}

// ---------------------------------------------------------------------------
// Monotonic alignment search: one wave32 per batch. Lane l owns columns
// j = l + 32c (c<12). Forward DP is row-parallel (only needs prev row);
// take-prev bits captured via __ballot into LDS; backtrack reads LDS bits.
__global__ __launch_bounds__(32)
void k_mas(const float* __restrict__ soft, float* __restrict__ hard,
           float* __restrict__ dur) {
  __shared__ unsigned int bits[TDLEN][12];   // 48 KB
  __shared__ unsigned short currArr[TDLEN];  // 2 KB
  __shared__ float sdur[TELEN];
  int l = threadIdx.x;
  int b = blockIdx.x;
  const float* S = soft + (size_t)b * TDLEN * TELEN;

  float logp[12];
#pragma unroll
  for (int c = 0; c < 12; ++c) {
    int j = l + 32 * c;
    logp[c] = (j == 0) ? logf(S[0]) : -__builtin_inff();
  }
  for (int i = 1; i < TDLEN; ++i) {
    float la[12];
#pragma unroll
    for (int c = 0; c < 12; ++c) la[c] = logf(S[(size_t)i * TELEN + l + 32 * c]);
    float newlp[12];
#pragma unroll
    for (int c = 0; c < 12; ++c) {
      float up = __shfl_up(logp[c], 1);
      float lastPrev = __shfl(logp[c > 0 ? c - 1 : 0], 31);
      float shifted = (l == 0) ? ((c == 0) ? -__builtin_inff() : lastPrev) : up;
      bool take = (shifted >= logp[c]) && !((l == 0) && (c == 0));
      unsigned long long bal = __ballot(take);
      if (l == c) bits[i][c] = (unsigned int)bal;
      newlp[c] = la[c] + (take ? shifted : logp[c]);
    }
#pragma unroll
    for (int c = 0; c < 12; ++c) logp[c] = newlp[c];
  }
  __syncthreads();

  if (l == 0) {
    int curr = TELEN - 1;
    for (int i = TDLEN - 1; i >= 1; --i) {
      currArr[i] = (unsigned short)curr;
      unsigned int w = bits[i][curr >> 5];
      if ((w >> (curr & 31)) & 1u) curr -= 1;
    }
    currArr[0] = (unsigned short)curr;
  }
#pragma unroll
  for (int c = 0; c < 12; ++c) sdur[l + 32 * c] = 0.f;
  __syncthreads();

  float* H = hard + (size_t)b * TDLEN * TELEN;
  for (int i = l; i < TDLEN; i += 32) {
    int cu = currArr[i];
    H[(size_t)i * TELEN + cu] = 1.0f;
    atomicAdd(&sdur[cu], 1.0f);
  }
  if (l == 0 && currArr[0] != 0) {   // replicate opt.at[0,0].set(1)
    H[0] = 1.0f;
    atomicAdd(&sdur[0], 1.0f);
  }
  __syncthreads();
#pragma unroll
  for (int c = 0; c < 12; ++c) dur[(size_t)b * TELEN + l + 32 * c] = sdur[l + 32 * c];
}

// ---------------------------------------------------------------------------
extern "C" void kernel_launch(void* const* d_in, const int* in_sizes, int n_in,
                              void* d_out, int out_size, void* d_ws, size_t ws_size,
                              hipStream_t stream) {
  const float* enc_in = (const float*)d_in[0];
  const float* dec_in = (const float*)d_in[1];
  const float* prior  = (const float*)d_in[5];
  const float* kw1 = (const float*)d_in[6];  const float* kb1 = (const float*)d_in[7];
  const float* kw2 = (const float*)d_in[8];  const float* kb2 = (const float*)d_in[9];
  const float* qw1 = (const float*)d_in[10]; const float* qb1 = (const float*)d_in[11];
  const float* qw2 = (const float*)d_in[12]; const float* qb2 = (const float*)d_in[13];
  const float* qw3 = (const float*)d_in[14]; const float* qb3 = (const float*)d_in[15];

  char* ws = (char*)d_ws;
  // Halo'd activation buffers: Tp = T + 2*pad rows per batch, interior at +pad rows.
  _Float16* enc16  = (_Float16*)(ws + 0);                 // 12,648,448 B [32][386][512]
  _Float16* hidE1  = (_Float16*)(ws + 12648448);          //  6,291,456 B [32][384][256]
  _Float16* hidQ2  = (_Float16*)(ws + 0);                 // alias (16.78MB <= 18.94MB)
  _Float16* dec16  = (_Float16*)(ws + 18939904);          //  6,328,320 B [32][1030][96]
  _Float16* hidQ1  = (_Float16*)(ws + 25268224);          // 16,875,520 B [32][1030][256]
  _Float16* keys16 = (_Float16*)(ws + 42143744);          //  2,359,296 B [32][384][96]
  _Float16* q16    = (_Float16*)(ws + 44503040);          //  6,291,456 B [32][1024][96]
  float*    k2     = (float*)   (ws + 50794496);          //     49,152 B
  float*    q2     = (float*)   (ws + 50843648);          //    131,072 B
  _Float16* wE1    = (_Float16*)(ws + 50974720);          //    786,432 B
  _Float16* wE2    = (_Float16*)(ws + 51761152);          //     40,960 B
  _Float16* wQ1    = (_Float16*)(ws + 51802112);          //    344,064 B
  _Float16* wQ2    = (_Float16*)(ws + 52146176);          //    917,504 B
  _Float16* wQ3    = (_Float16*)(ws + 53063680);          //     40,960 B

  float* out     = (float*)d_out;
  float* logprob = out;
  float* soft    = out + (size_t)NB * TDLEN * TELEN;
  float* hard    = out + 2 * (size_t)NB * TDLEN * TELEN;
  float* dur     = out + 3 * (size_t)NB * TDLEN * TELEN;

  // zero halos + pad channels + sq accumulators + attn_hard (before producers)
  hipMemsetAsync(enc16,  0, 12648448, stream);
  hipMemsetAsync(dec16,  0, 6328320,  stream);
  hipMemsetAsync(hidQ1,  0, 16875520, stream);
  hipMemsetAsync(keys16, 0, 2359296,  stream);
  hipMemsetAsync(q16,    0, 6291456,  stream);
  hipMemsetAsync(k2,     0, 49152,    stream);
  hipMemsetAsync(q2,     0, 131072,   stream);
  hipMemsetAsync(hard,   0, (size_t)NB * TDLEN * TELEN * 4, stream);

  // weight repack (f32 -> f16, (dk,ci) K-order, padded Cin stride)
  k_prep_weights<<<dim3(1536), dim3(256), 0, stream>>>(kw1, wE1, 256, 512, 512, 3);
  k_prep_weights<<<dim3(80),   dim3(256), 0, stream>>>(kw2, wE2,  80, 256, 256, 1);
  k_prep_weights<<<dim3(672),  dim3(256), 0, stream>>>(qw1, wQ1, 256,  80,  96, 7);
  k_prep_weights<<<dim3(1792), dim3(256), 0, stream>>>(qw2, wQ2, 256, 256, 256, 7);
  k_prep_weights<<<dim3(80),   dim3(256), 0, stream>>>(qw3, wQ3,  80, 256, 256, 1);

  // activations to [T][C] f16 interiors
  k_transpose16<<<dim3(768, NB), dim3(256), 0, stream>>>(enc_in, enc16 + 512,
                                                         512, TELEN, 512, 386 * 512);
  k_transpose16<<<dim3(320, NB), dim3(256), 0, stream>>>(dec_in, dec16 + 3 * 96,
                                                         80, TDLEN, 96, 1030 * 96);

  // encoder path: 512 -k3-> 256 (leaky) -k1-> 80 (+k2 sums)
  k_conv32<<<dim3(12, NB, 8), dim3(32), 0, stream>>>(
      enc16 + 512, wE1, kb1, hidE1, 512, 3, 256, 386 * 512, 384 * 256);
  k_conv16_proj<<<dim3(24, NB, 5), dim3(32), 0, stream>>>(
      hidE1, wE2, kb2, keys16, k2, TELEN, 256, CPAD, 384 * 256, 384 * CPAD);

  // decoder path: 80(->96) -k7-> 256 (leaky) -k7-> 256 (leaky) -k1-> 80 (+q2 sums)
  k_conv32<<<dim3(32, NB, 8), dim3(32), 0, stream>>>(
      dec16 + 3 * 96, wQ1, qb1, hidQ1 + 3 * 256, 96, 7, 256, 1030 * 96, 1030 * 256);
  k_conv32<<<dim3(32, NB, 8), dim3(32), 0, stream>>>(
      hidQ1 + 3 * 256, wQ2, qb2, hidQ2, 256, 7, 256, 1030 * 256, 1024 * 256);
  k_conv16_proj<<<dim3(64, NB, 5), dim3(32), 0, stream>>>(
      hidQ2, wQ3, qb3, q16, q2, TDLEN, 256, CPAD, 1024 * 256, 1024 * CPAD);

  // scores (WMMA QK^T, K=96) + log_softmax + prior + softmax
  k_scores<<<dim3(64, NB), dim3(256), 0, stream>>>(q16, keys16, q2, k2, prior,
                                                   logprob, soft);
  // monotonic alignment search + hard attention + durations
  k_mas<<<dim3(NB), dim3(32), 0, stream>>>(soft, hard, dur);
}
